// MambaEncoder_6107443495371
// MI455X (gfx1250) — compile-verified, matrix operand order checked by reading, not run on hardware
//
#include <hip/hip_runtime.h>
#include <hip/hip_bf16.h>
#include <math.h>

// ---------------- problem constants (match reference) ----------------
#define kB   8
#define kL   512
#define kFIN 128
#define kDM  256     // D_MODEL
#define kDI  512     // D_INNER
#define kDS  16      // D_STATE
#define kDTR 16      // DT_RANK
#define kDC  4       // D_CONV
#define kNL  3
#define kT   (kB * kL)   // 4096 tokens

typedef __attribute__((ext_vector_type(16))) _Float16 v16h;
typedef __attribute__((ext_vector_type(8)))  _Float16 v8h;
typedef __attribute__((ext_vector_type(8)))  float    v8f;

__device__ __forceinline__ float dev_silu(float v)     { return v / (1.f + __expf(-v)); }
__device__ __forceinline__ float dev_softplus(float v) { return (v > 20.f) ? v : log1pf(__expf(v)); }

template<int ACT>
__device__ __forceinline__ float dev_act(float v)
{
    if (ACT == 1) return dev_silu(v);
    if (ACT == 2) return dev_softplus(v);
    return v;
}

// A fragment (16x32 f16, CDNA5 layout): lane holds row=lane%16,
// k = kb..kb+7 (elems 0-7) and kb+16..kb+23 (elems 8-15), kb = 8*(lane/16).
__device__ __forceinline__ v16h load_a_frag_fast(const float* __restrict__ Arow, int kb)
{
    const float4 q0 = *(const float4*)(Arow + kb);
    const float4 q1 = *(const float4*)(Arow + kb + 4);
    const float4 q2 = *(const float4*)(Arow + kb + 16);
    const float4 q3 = *(const float4*)(Arow + kb + 20);
    v16h f;
    f[0]  = (_Float16)q0.x; f[1]  = (_Float16)q0.y; f[2]  = (_Float16)q0.z; f[3]  = (_Float16)q0.w;
    f[4]  = (_Float16)q1.x; f[5]  = (_Float16)q1.y; f[6]  = (_Float16)q1.z; f[7]  = (_Float16)q1.w;
    f[8]  = (_Float16)q2.x; f[9]  = (_Float16)q2.y; f[10] = (_Float16)q2.z; f[11] = (_Float16)q2.w;
    f[12] = (_Float16)q3.x; f[13] = (_Float16)q3.y; f[14] = (_Float16)q3.z; f[15] = (_Float16)q3.w;
    return f;
}

template<int ACT, bool HB, bool HC>
__device__ __forceinline__ void store_tile(float* __restrict__ C,
                                           const float* __restrict__ bias,
                                           const float* __restrict__ Cadd,
                                           v8f acc, int rbase, int col, int ldc, int half)
{
    const float bv = HB ? bias[col] : 0.f;
#pragma unroll
    for (int j = 0; j < 8; ++j) {
        const int rowm = rbase + j + half * 8;
        float v = dev_act<ACT>(acc[j] + bv);
        const size_t idx = (size_t)rowm * ldc + col;
        if (HC) v += Cadd[idx];
        C[idx] = v;
    }
}

#define kBSTRIDE 40   // padded LDS row stride (halves): 16B-aligned, bank-spread

// =====================================================================
// FAST WMMA GEMM: requires M%256==0, N%32==0, K%32==0.
// Block = 256 rows x 32 cols (8 waves, each a 32x32 tile sharing one
// LDS-staged B tile). Coalesced float4 B loads + transposed f16 LDS
// staging; per-lane fragment reads are two 16B LDS vector loads.
// =====================================================================
template<int ACT, bool HB, bool HC>
__global__ __launch_bounds__(256, 1)
void mamba_gemm_fast(const float* __restrict__ A,
                     const float* __restrict__ Bm,
                     const float* __restrict__ bias,
                     const float* __restrict__ Cadd,
                     float* __restrict__ C,
                     int M, int N, int K, int lda, int ldb, int ldc)
{
    __shared__ _Float16 Bs[32][kBSTRIDE];              // [n_local][k_local]

    const int wv     = threadIdx.x >> 5;               // 0..7 wave in block
    const int lane   = threadIdx.x & 31;
    const int tilesN = N >> 5;
    const int rowBlk = blockIdx.x / tilesN;
    const int colBlk = blockIdx.x % tilesN;
    const int row0   = (rowBlk * 8 + wv) << 5;
    const int col0   = colBlk << 5;
    const int r      = lane & 15;
    const int half   = lane >> 4;

    const float* Arow0 = A + (size_t)(row0 + r) * lda;
    const float* Arow1 = Arow0 + (size_t)16 * lda;

    // cooperative B staging: thread -> (k = tid/8, n4 = 4*(tid%8))
    const int tk  = threadIdx.x >> 3;                  // 0..31
    const int tn4 = (threadIdx.x & 7) << 2;            // 0,4,...,28
    const float* Bp = Bm + (size_t)tk * ldb + col0 + tn4;

    v8f acc00 = {}, acc01 = {}, acc10 = {}, acc11 = {};
#pragma unroll 1
    for (int k0 = 0; k0 < K; k0 += 32) {
        // ---- stage B tile (32k x 32n), transposed into Bs[n][k] ----
        const float4 q = *(const float4*)(Bp + (size_t)k0 * ldb);
        Bs[tn4 + 0][tk] = (_Float16)q.x;
        Bs[tn4 + 1][tk] = (_Float16)q.y;
        Bs[tn4 + 2][tk] = (_Float16)q.z;
        Bs[tn4 + 3][tk] = (_Float16)q.w;
        __syncthreads();

        const int kb = half * 8;                       // local k base for this lane
        v16h b0, b1;
        {
            const v8h lo0 = *(const v8h*)&Bs[r][kb];
            const v8h hi0 = *(const v8h*)&Bs[r][kb + 16];
            const v8h lo1 = *(const v8h*)&Bs[r + 16][kb];
            const v8h hi1 = *(const v8h*)&Bs[r + 16][kb + 16];
#pragma unroll
            for (int e = 0; e < 8; ++e) {
                b0[e] = lo0[e]; b0[e + 8] = hi0[e];
                b1[e] = lo1[e]; b1[e + 8] = hi1[e];
            }
        }

        if (k0 + 32 < K) {                             // prefetch next K slice of A
            __builtin_prefetch(Arow0 + k0 + kb + 32, 0, 3);
            __builtin_prefetch(Arow1 + k0 + kb + 32, 0, 3);
        }
        {
            const v16h a0 = load_a_frag_fast(Arow0, k0 + kb);
            acc00 = __builtin_amdgcn_wmma_f32_16x16x32_f16(false, a0, false, b0, (short)0, acc00, false, false);
            acc01 = __builtin_amdgcn_wmma_f32_16x16x32_f16(false, a0, false, b1, (short)0, acc01, false, false);
        }
        {
            const v16h a1 = load_a_frag_fast(Arow1, k0 + kb);
            acc10 = __builtin_amdgcn_wmma_f32_16x16x32_f16(false, a1, false, b0, (short)0, acc10, false, false);
            acc11 = __builtin_amdgcn_wmma_f32_16x16x32_f16(false, a1, false, b1, (short)0, acc11, false, false);
        }
        __syncthreads();                               // Bs reuse barrier
    }
    const int n0 = col0 + r, n1 = col0 + 16 + r;
    store_tile<ACT, HB, HC>(C, bias, Cadd, acc00, row0,      n0, ldc, half);
    store_tile<ACT, HB, HC>(C, bias, Cadd, acc01, row0,      n1, ldc, half);
    store_tile<ACT, HB, HC>(C, bias, Cadd, acc10, row0 + 16, n0, ldc, half);
    store_tile<ACT, HB, HC>(C, bias, Cadd, acc11, row0 + 16, n1, ldc, half);
}

// =====================================================================
// GENERIC WMMA GEMM (branchless guards on K and N; M%16==0 assumed).
// One wave32 per 16x16 tile.
// =====================================================================
template<int ACT, bool HB, bool HC>
__global__ __launch_bounds__(256, 1)
void mamba_gemm_gen(const float* __restrict__ A,
                    const float* __restrict__ Bm,
                    const float* __restrict__ bias,
                    const float* __restrict__ Cadd,
                    float* __restrict__ C,
                    int M, int N, int K, int lda, int ldb, int ldc)
{
    const int wave   = blockIdx.x * (blockDim.x >> 5) + (threadIdx.x >> 5);
    const int lane   = threadIdx.x & 31;
    const int tilesN = (N + 15) >> 4;
    if (wave >= (M >> 4) * tilesN) return;             // wave-uniform exit

    const int row0 = (wave / tilesN) << 4;
    const int col0 = (wave % tilesN) << 4;
    const int r    = lane & 15;
    const int half = lane >> 4;

    const float* Arow = A + (size_t)(row0 + r) * lda;
    const int  n   = col0 + r;
    const int  nn  = (n < N) ? n : 0;                  // clamped, always in-bounds
    const bool nok = (n < N);

    v8f acc = {};
#pragma unroll 1
    for (int k0 = 0; k0 < K; k0 += 32) {
        const int kb = k0 + half * 8;
        v16h af, bf;
#pragma unroll
        for (int e = 0; e < 8; ++e) {
            const int k1 = kb + e;
            const int k2 = kb + 16 + e;
            const int k1c = (k1 < K) ? k1 : 0;         // clamp -> unconditional load
            const int k2c = (k2 < K) ? k2 : 0;
            const float va1 = Arow[k1c], va2 = Arow[k2c];
            af[e]     = (_Float16)((k1 < K) ? va1 : 0.f);
            af[e + 8] = (_Float16)((k2 < K) ? va2 : 0.f);
            const float vb1 = Bm[(size_t)k1c * ldb + nn];
            const float vb2 = Bm[(size_t)k2c * ldb + nn];
            bf[e]     = (_Float16)((nok && k1 < K) ? vb1 : 0.f);
            bf[e + 8] = (_Float16)((nok && k2 < K) ? vb2 : 0.f);
        }
        acc = __builtin_amdgcn_wmma_f32_16x16x32_f16(false, af, false, bf, (short)0, acc, false, false);
    }
    if (nok) {
        const float bv = HB ? bias[n] : 0.f;
#pragma unroll
        for (int j = 0; j < 8; ++j) {
            const int rowm = row0 + j + half * 8;
            float v = dev_act<ACT>(acc[j] + bv);
            const size_t idx = (size_t)rowm * ldc + n;
            if (HC) v += Cadd[idx];
            C[idx] = v;
        }
    }
}

// ---------------- masked per-feature mean (imputer) -------------------
__global__ void mamba_featmean(const float* __restrict__ x,
                               const float* __restrict__ mask,
                               float* __restrict__ xmean)
{
    const int f = blockIdx.x, tid = threadIdx.x;
    __shared__ float rs[256], rc[256];
    float s = 0.f, c = 0.f;
    for (int t = tid; t < kT; t += 256) {
        const float mv = mask[(size_t)t * kFIN + f];
        s += x[(size_t)t * kFIN + f] * mv;
        c += mv;
    }
    rs[tid] = s; rc[tid] = c; __syncthreads();
    for (int st = 128; st > 0; st >>= 1) {
        if (tid < st) { rs[tid] += rs[tid + st]; rc[tid] += rc[tid + st]; }
        __syncthreads();
    }
    if (tid == 0) xmean[f] = rs[0] / fmaxf(rc[0], 1.f);
}

// ------- imputer + time-embedding elementwise preparation -------------
__global__ void mamba_token_prep(const float* __restrict__ x,
                                 const float* __restrict__ tau,
                                 const float* __restrict__ mask,
                                 const float* __restrict__ t_abs,
                                 const float* __restrict__ imp_log_w,
                                 const float* __restrict__ xmean,
                                 const float* __restrict__ gap_W1,
                                 const float* __restrict__ gap_b1,
                                 const float* __restrict__ gap_b2,
                                 const float* __restrict__ circ_W,
                                 float* __restrict__ x_hat,
                                 float* __restrict__ silu_vec,
                                 float* __restrict__ te_init)
{
    const int t   = blockIdx.x;
    const int tid = threadIdx.x;                       // 256
    const float tau_h = fmaxf(tau[t] / 3600.f, 1e-4f);
    const float g     = log1pf(tau_h);
    const float ta    = t_abs[t];
    const float hour  = fmodf(ta / 3600.f, 24.f);
    const float dow   = fmodf(ta / 86400.f, 7.f);
    const float twopi = 6.283185307179586f;
    const float c0 = sinf(twopi * hour / 24.f);
    const float c1 = cosf(twopi * hour / 24.f);
    const float c2 = sinf(twopi * dow / 7.f);
    const float c3 = cosf(twopi * dow / 7.f);

    if (tid < kFIN) {
        const int f = tid;
        const float rate  = dev_softplus(imp_log_w[f]);
        const float gamma = __expf(-rate * tau_h);
        const float xv = x[(size_t)t * kFIN + f];
        const float mv = mask[(size_t)t * kFIN + f];
        x_hat[(size_t)t * kFIN + f] =
            mv * xv + (1.f - mv) * (gamma * xv + (1.f - gamma) * xmean[f]);
    }
    const int d = tid;                                 // 0..255
    silu_vec[(size_t)t * kDM + d] = dev_silu(g * gap_W1[d] + gap_b1[d]);
    te_init[(size_t)t * kDM + d] =
        gap_b2[d] + c0 * circ_W[0 * kDM + d] + c1 * circ_W[1 * kDM + d]
                  + c2 * circ_W[2 * kDM + d] + c3 * circ_W[3 * kDM + d];
}

// -------------- rowwise layernorm over 256 cols -----------------------
__global__ void mamba_layernorm(const float* __restrict__ in,
                                float* __restrict__ out,
                                const float* __restrict__ w,
                                const float* __restrict__ b,
                                int sanitize)
{
    const int row = blockIdx.x, c = threadIdx.x;       // 256 threads
    __shared__ float red[256];
    __shared__ float s_mean, s_rstd;
    const float v = in[(size_t)row * kDM + c];
    red[c] = v; __syncthreads();
    for (int st = 128; st > 0; st >>= 1) {
        if (c < st) red[c] += red[c + st];
        __syncthreads();
    }
    if (c == 0) s_mean = red[0] / (float)kDM;
    __syncthreads();
    const float m = s_mean;
    const float dv = v - m;
    red[c] = dv * dv; __syncthreads();
    for (int st = 128; st > 0; st >>= 1) {
        if (c < st) red[c] += red[c + st];
        __syncthreads();
    }
    if (c == 0) s_rstd = rsqrtf(red[0] / (float)kDM + 1e-5f);
    __syncthreads();
    float o = dv * s_rstd * w[c] + b[c];
    if (sanitize) {
        if (isnan(o)) o = 0.f;
        else if (isinf(o)) o = (o > 0.f) ? 1.f : -1.f;
    }
    out[(size_t)row * kDM + c] = o;
}

// ---------------- concat(proj, te) -> fuse input ----------------------
__global__ void mamba_concat(const float* __restrict__ proj,
                             const float* __restrict__ te,
                             float* __restrict__ fuse_in)
{
    for (size_t idx = blockIdx.x * blockDim.x + threadIdx.x;
         idx < (size_t)kT * 2 * kDM; idx += (size_t)gridDim.x * blockDim.x) {
        const size_t t = idx / (2 * kDM);
        const int    j = (int)(idx % (2 * kDM));
        fuse_in[idx] = (j < kDM) ? proj[t * kDM + j] : te[t * kDM + (j - kDM)];
    }
}

// ---------- causal depthwise conv (width 4) + SiLU --------------------
__global__ void mamba_conv_silu(const float* __restrict__ xz,
                                const float* __restrict__ convW,  // [kDI,4]
                                const float* __restrict__ convb,  // [kDI]
                                float* __restrict__ xc)
{
    for (size_t idx = blockIdx.x * blockDim.x + threadIdx.x;
         idx < (size_t)kT * kDI; idx += (size_t)gridDim.x * blockDim.x) {
        const int d = (int)(idx % kDI);
        const int l = (int)((idx / kDI) % kL);
        const int b = (int)(idx / ((size_t)kDI * kL));
        float s = convb[d];
#pragma unroll
        for (int j = 0; j < kDC; ++j) {
            const int li = l - (kDC - 1) + j;
            if (li >= 0)
                s += convW[d * kDC + j] *
                     xz[((size_t)(b * kL + li)) * (2 * kDI) + d];
        }
        xc[idx] = dev_silu(s);
    }
}

// ------------- selective scan: 1 block per batch, 1 lane/channel ------
__global__ void mamba_scan(const float* __restrict__ dt,      // [T,kDI]
                           const float* __restrict__ x_dbl,   // [T,48] (dt|B|C)
                           const float* __restrict__ xc,      // [T,kDI]
                           const float* __restrict__ xz,      // [T,2*kDI]
                           const float* __restrict__ A_log,   // [kDI,16]
                           const float* __restrict__ Dp,      // [kDI]
                           float* __restrict__ y)             // [T,kDI]
{
    const int b = blockIdx.x;
    const int d = threadIdx.x;                         // 512 threads = 16 waves
    __shared__ float sBC[2 * kDS];

    float Aneg[kDS];
#pragma unroll
    for (int k = 0; k < kDS; ++k) Aneg[k] = -__expf(A_log[(size_t)d * kDS + k]);

    float s[kDS];
#pragma unroll
    for (int k = 0; k < kDS; ++k) s[k] = 0.f;

    for (int l = 0; l < kL; ++l) {
        const size_t t = (size_t)b * kL + l;
        if (d < 2 * kDS) sBC[d] = x_dbl[t * (kDTR + 2 * kDS) + kDTR + d];
        __syncthreads();

        const float dtv = dt[t * kDI + d];
        const float xcv = xc[t * kDI + d];
        float yv = 0.f;
#pragma unroll
        for (int k = 0; k < kDS; ++k) {
            s[k] = s[k] * __expf(dtv * Aneg[k]) + dtv * sBC[k] * xcv;
            yv  += s[k] * sBC[kDS + k];
        }
        const float zv = xz[t * (2 * kDI) + kDI + d];
        y[t * kDI + d] = (yv + xcv * Dp[d]) * dev_silu(zv);
        __syncthreads();
    }
}

// -------------- residual: h += m * exp(-softplus(decay)*tau_h) --------
__global__ void mamba_residual(const float* __restrict__ m,
                               const float* __restrict__ tau,
                               const float* __restrict__ decay,
                               float* __restrict__ h)
{
    for (size_t idx = blockIdx.x * blockDim.x + threadIdx.x;
         idx < (size_t)kT * kDM; idx += (size_t)gridDim.x * blockDim.x) {
        const size_t t = idx / kDM;
        const int    d = (int)(idx % kDM);
        const float th = fmaxf(tau[t] / 3600.f, 1e-4f);
        h[idx] += m[idx] * __expf(-dev_softplus(decay[d]) * th);
    }
}

// -------------------------- pooling -----------------------------------
__global__ void mamba_valid_last(const float* __restrict__ mask,
                                 float* __restrict__ valid,
                                 float* __restrict__ nobs,
                                 int* __restrict__ lastidx)
{
    const int b = blockIdx.x, tid = threadIdx.x;       // 256
    __shared__ float rs[256]; __shared__ int rm[256];
    float cnt = 0.f; int lmax = -1;
    for (int l = tid; l < kL; l += 256) {
        const float* mp = mask + ((size_t)(b * kL + l)) * kFIN;
        float mx = 0.f;
        for (int f = 0; f < kFIN; ++f) mx = fmaxf(mx, mp[f]);
        const float v = (mx > 0.f) ? 1.f : 0.f;
        valid[b * kL + l] = v;
        cnt += v;
        if (v > 0.f && l > lmax) lmax = l;
    }
    rs[tid] = cnt; rm[tid] = lmax; __syncthreads();
    for (int st = 128; st > 0; st >>= 1) {
        if (tid < st) { rs[tid] += rs[tid + st]; rm[tid] = max(rm[tid], rm[tid + st]); }
        __syncthreads();
    }
    if (tid == 0) { nobs[b] = fmaxf(rs[0], 1.f); lastidx[b] = max(rm[0], 0); }
}

__global__ void mamba_pool_stats(const float* __restrict__ hfin,
                                 const float* __restrict__ valid,
                                 const float* __restrict__ nobs,
                                 const int* __restrict__ lastidx,
                                 float* __restrict__ zcat)
{
    const int idx = blockIdx.x * blockDim.x + threadIdx.x;
    if (idx >= kB * kDM) return;
    const int b = idx / kDM, d = idx % kDM;
    float sum = 0.f, mx = -3.4e38f;
    for (int l = 0; l < kL; ++l) {
        const float hv = hfin[((size_t)(b * kL + l)) * kDM + d];
        sum += hv * valid[b * kL + l];
        mx = fmaxf(mx, hv);
    }
    float* z = zcat + (size_t)b * 4 * kDM;
    z[d]            = hfin[((size_t)(b * kL + lastidx[b])) * kDM + d];
    z[kDM + d]      = sum / nobs[b];
    z[2 * kDM + d]  = mx;
}

__global__ void mamba_scores(const float* __restrict__ hfin,
                             const float* __restrict__ poolW,   // [kDM, kDM/2]
                             const float* __restrict__ poolb,
                             const float* __restrict__ poolu,
                             const float* __restrict__ valid,
                             float* __restrict__ scores)
{
    const int t = blockIdx.x, j = threadIdx.x;         // 128 threads
    __shared__ float red[128];
    float acc = poolb[j];
    const float* hp = hfin + (size_t)t * kDM;
    for (int c = 0; c < kDM; ++c) acc += hp[c] * poolW[c * (kDM / 2) + j];
    red[j] = tanhf(acc) * poolu[j];
    __syncthreads();
    for (int st = 64; st > 0; st >>= 1) {
        if (j < st) red[j] += red[j + st];
        __syncthreads();
    }
    if (j == 0) scores[t] = (valid[t] > 0.f) ? red[0] : -__builtin_inff();
}

__global__ void mamba_attn(const float* __restrict__ scores,
                           const float* __restrict__ hfin,
                           float* __restrict__ zcat)
{
    const int b = blockIdx.x, tid = threadIdx.x;       // 256
    __shared__ float alpha[kL];
    __shared__ float red[256];
    __shared__ float s_mx, s_z;
    float mx = -3.4e38f; bool any = false;
    for (int l = tid; l < kL; l += 256) {
        const float sc = scores[b * kL + l];
        if (!isinf(sc)) { mx = fmaxf(mx, sc); any = true; }
    }
    red[tid] = any ? mx : -3.4e38f; __syncthreads();
    for (int st = 128; st > 0; st >>= 1) {
        if (tid < st) red[tid] = fmaxf(red[tid], red[tid + st]);
        __syncthreads();
    }
    if (tid == 0) s_mx = red[0];
    __syncthreads();
    const float gmx = s_mx;
    const bool uniform = (gmx <= -3.0e38f);
    float ssum = 0.f;
    for (int l = tid; l < kL; l += 256) {
        const float sc = scores[b * kL + l];
        const float a  = uniform ? (1.f / kL)
                                 : (isinf(sc) ? 0.f : __expf(sc - gmx));
        alpha[l] = a; ssum += a;
    }
    red[tid] = ssum; __syncthreads();
    for (int st = 128; st > 0; st >>= 1) {
        if (tid < st) red[tid] += red[tid + st];
        __syncthreads();
    }
    if (tid == 0) s_z = red[0];
    __syncthreads();
    const float zsum = s_z;
    for (int l = tid; l < kL; l += 256) alpha[l] /= zsum;
    __syncthreads();
    const int d = tid;
    float acc = 0.f;
    for (int l = 0; l < kL; ++l)
        acc += alpha[l] * hfin[((size_t)(b * kL + l)) * kDM + d];
    zcat[(size_t)b * 4 * kDM + 3 * kDM + d] = acc;
}

__global__ void mamba_outproj(const float* __restrict__ zcat,
                              const float* __restrict__ projW,
                              const float* __restrict__ projb,
                              float* __restrict__ out)
{
    const int idx = blockIdx.x * blockDim.x + threadIdx.x;
    if (idx >= kB * kDM) return;
    const int b = idx / kDM, d = idx % kDM;
    float acc = projb[d];
    const float* z = zcat + (size_t)b * 4 * kDM;
    for (int j = 0; j < 4 * kDM; ++j) acc += z[j] * projW[j * kDM + d];
    out[idx] = acc;
}

// =====================================================================
// host side
// =====================================================================
template<int ACT, bool HB, bool HC>
static void launch_gemm(const float* A, const float* Bm, const float* bias,
                        const float* Cadd, float* C,
                        int M, int N, int K, int lda, int ldb, int ldc,
                        hipStream_t s)
{
    if ((M % 256) == 0 && (N % 32) == 0 && (K % 32) == 0) {
        const int blocks = (M / 256) * (N / 32);       // exact grid (barriers inside)
        hipLaunchKernelGGL((mamba_gemm_fast<ACT, HB, HC>), dim3(blocks), dim3(256), 0, s,
                           A, Bm, bias, Cadd, C, M, N, K, lda, ldb, ldc);
    } else {
        const int waves  = (M / 16) * ((N + 15) / 16);
        const int blocks = (waves + 7) / 8;
        hipLaunchKernelGGL((mamba_gemm_gen<ACT, HB, HC>), dim3(blocks), dim3(256), 0, s,
                           A, Bm, bias, Cadd, C, M, N, K, lda, ldb, ldc);
    }
}

extern "C" void kernel_launch(void* const* d_in, const int* in_sizes, int n_in,
                              void* d_out, int out_size, void* d_ws, size_t ws_size,
                              hipStream_t stream)
{
    (void)in_sizes; (void)n_in; (void)out_size; (void)ws_size;

    const float* x        = (const float*)d_in[0];
    const float* tau      = (const float*)d_in[1];
    const float* mask     = (const float*)d_in[2];
    const float* t_abs    = (const float*)d_in[3];
    const float* imp_logw = (const float*)d_in[4];
    const float* in_W     = (const float*)d_in[5];
    const float* in_b     = (const float*)d_in[6];
    const float* ln_in_w  = (const float*)d_in[7];
    const float* ln_in_b  = (const float*)d_in[8];
    const float* gap_W1   = (const float*)d_in[9];
    const float* gap_b1   = (const float*)d_in[10];
    const float* gap_W2   = (const float*)d_in[11];
    const float* gap_b2   = (const float*)d_in[12];
    const float* circ_W   = (const float*)d_in[13];
    const float* fuse_W   = (const float*)d_in[14];
    const float* fuse_b   = (const float*)d_in[15];
    const float* blk_ln_w = (const float*)d_in[16];
    const float* blk_ln_b = (const float*)d_in[17];
    const float* blk_dec  = (const float*)d_in[18];
    const float* in_projW = (const float*)d_in[19];
    const float* conv_W   = (const float*)d_in[20];
    const float* conv_b   = (const float*)d_in[21];
    const float* x_projW  = (const float*)d_in[22];
    const float* dt_W     = (const float*)d_in[23];
    const float* dt_b     = (const float*)d_in[24];
    const float* A_log    = (const float*)d_in[25];
    const float* Dparam   = (const float*)d_in[26];
    const float* out_W    = (const float*)d_in[27];
    const float* norm_w   = (const float*)d_in[28];
    const float* norm_b   = (const float*)d_in[29];
    const float* pool_W   = (const float*)d_in[30];
    const float* pool_Wb  = (const float*)d_in[31];
    const float* pool_u   = (const float*)d_in[32];
    const float* proj_W   = (const float*)d_in[33];
    const float* proj_b   = (const float*)d_in[34];
    float* out = (float*)d_out;

    // ---- workspace arena (floats) ----
    float* W = (float*)d_ws;
    size_t o = 0;
    float* xmean  = W + o; o += 128;
    float* valid  = W + o; o += kT;
    float* nobs   = W + o; o += 16;
    int*   lastix = (int*)(W + o); o += 16;
    float* scores = W + o; o += kT;
    float* zcat   = W + o; o += (size_t)kB * 4 * kDM;
    float* xhat   = W + o; o += (size_t)kT * kFIN;
    float* siluv  = W + o; o += (size_t)kT * kDM;     // later reused as hfin
    float* te     = W + o; o += (size_t)kT * kDM;
    float* proj   = W + o; o += (size_t)kT * kDM;
    float* h      = W + o; o += (size_t)kT * kDM;
    float* hn     = W + o; o += (size_t)kT * kDM;     // later reused as m
    float* xz     = W + o; o += (size_t)kT * 2 * kDI; // also fuse_in (phase B)
    float* xc     = W + o; o += (size_t)kT * kDI;
    float* xdbl   = W + o; o += (size_t)kT * (kDTR + 2 * kDS);
    float* dtb    = W + o; o += (size_t)kT * kDI;
    float* ybuf   = W + o; o += (size_t)kT * kDI;
    float* fuse_in = xz;
    float* mbuf    = hn;
    float* hfin    = siluv;

    // ================= phase A: imputer stats =================
    hipLaunchKernelGGL(mamba_featmean, dim3(kFIN), dim3(256), 0, stream, x, mask, xmean);

    // ================= phase B: embed + fuse =================
    hipLaunchKernelGGL(mamba_token_prep, dim3(kT), dim3(kDM), 0, stream,
                       x, tau, mask, t_abs, imp_logw, xmean,
                       gap_W1, gap_b1, gap_b2, circ_W, xhat, siluv, te);
    launch_gemm<0, true, false>(xhat, in_W, in_b, nullptr, proj,
                                kT, kDM, kFIN, kFIN, kDM, kDM, stream);
    hipLaunchKernelGGL(mamba_layernorm, dim3(kT), dim3(kDM), 0, stream,
                       proj, proj, ln_in_w, ln_in_b, 0);
    launch_gemm<0, false, true>(siluv, gap_W2, nullptr, te, te,
                                kT, kDM, kDM, kDM, kDM, kDM, stream);
    hipLaunchKernelGGL(mamba_concat, dim3(1024), dim3(256), 0, stream, proj, te, fuse_in);
    launch_gemm<0, true, false>(fuse_in, fuse_W, fuse_b, nullptr, h,
                                kT, kDM, 2 * kDM, 2 * kDM, kDM, kDM, stream);

    // ================= phase C: mamba layers =================
    for (int i = 0; i < kNL; ++i) {
        hipLaunchKernelGGL(mamba_layernorm, dim3(kT), dim3(kDM), 0, stream,
                           h, hn, blk_ln_w + i * kDM, blk_ln_b + i * kDM, 0);
        launch_gemm<0, false, false>(hn, in_projW + (size_t)i * kDM * 2 * kDI, nullptr, nullptr, xz,
                                     kT, 2 * kDI, kDM, kDM, 2 * kDI, 2 * kDI, stream);
        hipLaunchKernelGGL(mamba_conv_silu, dim3(2048), dim3(256), 0, stream,
                           xz, conv_W + (size_t)i * kDI * kDC, conv_b + (size_t)i * kDI, xc);
        launch_gemm<0, false, false>(xc, x_projW + (size_t)i * kDI * (kDTR + 2 * kDS), nullptr, nullptr, xdbl,
                                     kT, kDTR + 2 * kDS, kDI, kDI, kDTR + 2 * kDS, kDTR + 2 * kDS, stream);
        launch_gemm<2, true, false>(xdbl, dt_W + (size_t)i * kDTR * kDI, dt_b + (size_t)i * kDI, nullptr, dtb,
                                    kT, kDI, kDTR, kDTR + 2 * kDS, kDI, kDI, stream);
        hipLaunchKernelGGL(mamba_scan, dim3(kB), dim3(kDI), 0, stream,
                           dtb, xdbl, xc, xz,
                           A_log + (size_t)i * kDI * kDS, Dparam + (size_t)i * kDI, ybuf);
        launch_gemm<0, false, false>(ybuf, out_W + (size_t)i * kDI * kDM, nullptr, nullptr, mbuf,
                                     kT, kDM, kDI, kDI, kDM, kDM, stream);
        hipLaunchKernelGGL(mamba_residual, dim3(1024), dim3(256), 0, stream,
                           mbuf, tau, blk_dec + i * kDM, h);
    }

    // ================= phase D: final LN + pooling =================
    hipLaunchKernelGGL(mamba_layernorm, dim3(kT), dim3(kDM), 0, stream,
                       h, hfin, norm_w, norm_b, 1);
    hipLaunchKernelGGL(mamba_valid_last, dim3(kB), dim3(256), 0, stream,
                       mask, valid, nobs, lastix);
    hipLaunchKernelGGL(mamba_pool_stats, dim3((kB * kDM + 255) / 256), dim3(256), 0, stream,
                       hfin, valid, nobs, lastix, zcat);
    hipLaunchKernelGGL(mamba_scores, dim3(kT), dim3(kDM / 2), 0, stream,
                       hfin, pool_W, pool_Wb, pool_u, valid, scores);
    hipLaunchKernelGGL(mamba_attn, dim3(kB), dim3(256), 0, stream,
                       scores, hfin, zcat);
    hipLaunchKernelGGL(mamba_outproj, dim3((kB * kDM + 255) / 256), dim3(256), 0, stream,
                       zcat, proj_W, proj_b, out);
}
